// DifferentialAttention_80290118631914
// MI455X (gfx1250) — matrix operand order repaired and checked by reading, not compiled, not run on hardware
//
#include <hip/hip_runtime.h>

// ---------------------------------------------------------------------------
// Differential attention for MI455X (gfx1250), bf16 WMMA pipeline.
//   B=2, T=2048, D=2048, H=16, HD=128, HHD=64, SCALE=1/8
// Workspace layout (bytes), total 112 MB:
//   [0,16M)    x_bf16      (4096 x 2048)
//   [16M,24M)  Wq_bf16     (2048 x 2048)
//   [24M,32M)  Wk_bf16
//   [32M,40M)  Wv_bf16
//   [40M,48M)  Wo_bf16
//   [48M,64M)  Q_bf16      (4096 x 2048) row-major
//   [64M,80M)  K_bf16      (4096 x 2048) row-major
//   [80M,96M)  V^T_bf16    (2048 x 4096) channel-major (transposed GEMM store)
//   [96M,112M) attnO_bf16  (4096 x 2048)
// ---------------------------------------------------------------------------

typedef __attribute__((ext_vector_type(16))) __bf16 v16bf;
typedef __attribute__((ext_vector_type(8)))  float  v8f;

union BFrag { v16bf v; unsigned u[8]; uint4 q[2]; };
union CFrag { v8f   v; float    f[8]; };

__device__ __forceinline__ unsigned short f2bf(float x) {
    union { float f; unsigned u; } c; c.f = x;
    unsigned r = c.u + 0x7FFFu + ((c.u >> 16) & 1u);   // round-to-nearest-even
    return (unsigned short)(r >> 16);
}
__device__ __forceinline__ unsigned pack2bf(float a, float b) {
    return (unsigned)f2bf(a) | ((unsigned)f2bf(b) << 16);
}

// ---------------------------------------------------------------------------
// Kernel 1: fp32 -> bf16 streaming convert (vectorized: float4 -> 4x bf16)
// ---------------------------------------------------------------------------
__global__ __launch_bounds__(256) void cvt_f32_bf16(const float4* __restrict__ in,
                                                    uint2* __restrict__ out,
                                                    int n4) {
    int i = blockIdx.x * 256 + threadIdx.x;
    if (i < n4) {
        float4 v = in[i];
        uint2 o;
        o.x = pack2bf(v.x, v.y);
        o.y = pack2bf(v.z, v.w);
        out[i] = o;
    }
}

// ---------------------------------------------------------------------------
// Kernel 2: bf16 GEMM  C(MxN) = A(MxK) @ B(KxN), fp32 accumulate.
// BM=BN=128, BK=32, 256 threads (8 waves, 4x2 wave grid, 2x4 WMMA tiles/wave).
// Double-buffered LDS: one barrier/iter, global loads overlap WMMA compute.
// ---------------------------------------------------------------------------
#define BM 128
#define BN 128
#define BK 32

template <int Mi, int Ni, int Ki, bool OUTF32, bool TRANS>
__global__ __launch_bounds__(256) void gemm_bf16(
    const unsigned short* __restrict__ A,
    const unsigned short* __restrict__ Bm,
    void* __restrict__ Cp)
{
    __shared__ unsigned short As[2][BM * BK];   // row-major [row][k]   2x 8 KB
    __shared__ unsigned short Bs[2][BN * BK];   // transposed [col][k]  2x 8 KB

    const int tid   = threadIdx.x;
    const int lane  = tid & 31;
    const int wid   = tid >> 5;
    const int mlane = lane & 15;
    const int half  = lane >> 4;
    const int wm    = wid & 3;               // 4 wave-rows of 32
    const int wn    = wid >> 2;              // 2 wave-cols of 64
    const int row0  = blockIdx.y * BM;
    const int col0  = blockIdx.x * BN;

    CFrag acc[2][4];
#pragma unroll
    for (int rt = 0; rt < 2; ++rt)
#pragma unroll
        for (int ct = 0; ct < 4; ++ct)
#pragma unroll
            for (int i = 0; i < 8; ++i) acc[rt][ct].f[i] = 0.0f;

    // Staged global tile in registers.
    uint4 aval[2];
    uint2 b0val[2], b1val[2];

    auto load_tile = [&](int k0) {
#pragma unroll
        for (int c = 0; c < 2; ++c) {
            int linear = c * 256 + tid;              // 0..511
            int r  = linear >> 2;                    // A: 4 uint4 per row
            int cc = (linear & 3) * 8;
            aval[c] = *(const uint4*)&A[(size_t)(row0 + r) * Ki + k0 + cc];
            int colg = (linear & 31) * 4;            // B: column group of 4
            int kr   = (linear >> 5) * 2;            // even k row
            const unsigned short* bp = &Bm[(size_t)(k0 + kr) * Ni + col0 + colg];
            b0val[c] = *(const uint2*)bp;
            b1val[c] = *(const uint2*)(bp + Ni);
        }
    };
    auto store_tile = [&](int buf) {
#pragma unroll
        for (int c = 0; c < 2; ++c) {
            int linear = c * 256 + tid;
            int r  = linear >> 2;
            int cc = (linear & 3) * 8;
            *(uint4*)&As[buf][r * BK + cc] = aval[c];
            int colg = (linear & 31) * 4;
            int kr   = (linear >> 5) * 2;
            const unsigned short* s0 = (const unsigned short*)&b0val[c];
            const unsigned short* s1 = (const unsigned short*)&b1val[c];
#pragma unroll
            for (int e = 0; e < 4; ++e) {
                unsigned pk = (unsigned)s0[e] | ((unsigned)s1[e] << 16);
                *(unsigned*)&Bs[buf][(colg + e) * BK + kr] = pk;
            }
        }
    };

    load_tile(0);
    store_tile(0);
    constexpr int NK = Ki / BK;
    for (int i = 0; i < NK; ++i) {
        __syncthreads();                         // LDS buf (i&1) visible
        if (i + 1 < NK) load_tile((i + 1) * BK); // overlap global with compute
        const int buf = i & 1;

        // A fragments: lane=row(m); u[j]=K-pair 2j+16h (contiguous 16 elems)
        BFrag a[2];
#pragma unroll
        for (int rt = 0; rt < 2; ++rt) {
            const unsigned short* ab = &As[buf][(wm * 32 + rt * 16 + mlane) * BK + 16 * half];
            a[rt].q[0] = *(const uint4*)ab;
            a[rt].q[1] = *(const uint4*)(ab + 8);
        }
        // B fragments: lane=col(m); u[j]=K-pair 2j+16h (contiguous 16 elems)
        BFrag b[4];
#pragma unroll
        for (int ct = 0; ct < 4; ++ct) {
            const unsigned short* bb = &Bs[buf][(wn * 64 + ct * 16 + mlane) * BK + 16 * half];
            b[ct].q[0] = *(const uint4*)bb;
            b[ct].q[1] = *(const uint4*)(bb + 8);
        }
#pragma unroll
        for (int rt = 0; rt < 2; ++rt)
#pragma unroll
            for (int ct = 0; ct < 4; ++ct)
                acc[rt][ct].v = __builtin_amdgcn_wmma_f32_16x16x32_bf16(
                    false, a[rt].v, false, b[ct].v,
                    (short)0, acc[rt][ct].v, false, false);

        if (i + 1 < NK) store_tile((i + 1) & 1); // write the *other* buffer
    }

    // Store epilogue: C layout lane=(N=mlane), VGPR r -> M = r + 8*half.
#pragma unroll
    for (int rt = 0; rt < 2; ++rt) {
        const int grow0 = row0 + wm * 32 + rt * 16 + 8 * half;
#pragma unroll
        for (int ct = 0; ct < 4; ++ct) {
            const int gcol = col0 + wn * 64 + ct * 16 + mlane;
            if constexpr (OUTF32) {
                float* p = (float*)Cp + (size_t)grow0 * Ni + gcol;
#pragma unroll
                for (int r = 0; r < 8; ++r) p[(size_t)r * Ni] = acc[rt][ct].f[r];
            } else if constexpr (TRANS) {
                // 8 consecutive M values per lane -> single 16B store.
                unsigned short* p = (unsigned short*)Cp + (size_t)gcol * Mi + grow0;
                uint4 pk;
                pk.x = pack2bf(acc[rt][ct].f[0], acc[rt][ct].f[1]);
                pk.y = pack2bf(acc[rt][ct].f[2], acc[rt][ct].f[3]);
                pk.z = pack2bf(acc[rt][ct].f[4], acc[rt][ct].f[5]);
                pk.w = pack2bf(acc[rt][ct].f[6], acc[rt][ct].f[7]);
                *(uint4*)p = pk;
            } else {
                unsigned short* p = (unsigned short*)Cp + (size_t)grow0 * Ni + gcol;
#pragma unroll
                for (int r = 0; r < 8; ++r) p[(size_t)r * Ni] = f2bf(acc[rt][ct].f[r]);
            }
        }
    }
}

// ---------------------------------------------------------------------------
// Kernel 3: flash-style differential attention.
// grid = (B*H, T/128), block = 256 (8 waves). Each wave owns 16 query rows,
// streams causal key chunks of 32, online softmax for both score matrices,
// epilogue: O = O1/l1 - sigmoid(lambda)*O2/l2 -> bf16.
// All Q/K/V/P fragments load as 2x b128 (contiguous 16-element runs per lane).
// Overreads past the causal frontier stay inside d_ws and are masked/zeroed.
// ---------------------------------------------------------------------------
#define SCALE_QK 0.125f

template <int T_, int Hn, int HDn, int Bt>
__global__ __launch_bounds__(256) void diff_attn(
    const unsigned short* __restrict__ Qb,   // (B*T, H*HD)
    const unsigned short* __restrict__ Kb,   // (B*T, H*HD)
    const unsigned short* __restrict__ Vt,   // (H*HD, B*T) channel-major
    const float* __restrict__ lambda_init,   // (H,)
    unsigned short* __restrict__ Ob)         // (B*T, H*HD)
{
    constexpr int D2 = Hn * HDn;                // 2048
    constexpr int BT = Bt * T_;                 // 4096
    __shared__ unsigned short Plds[8 * 1024];   // per-wave: P1[16x32] P2[16x32]

    const int bh    = blockIdx.x;
    const int b     = bh / Hn, h = bh % Hn;
    const int lane  = threadIdx.x & 31;
    const int wid   = threadIdx.x >> 5;
    const int mlane = lane & 15;
    const int half  = lane >> 4;
    const int q0    = blockIdx.y * 128 + wid * 16;

    // Q fragments (A operand): 4 chunks of 32 channels (0-63 -> Q1, 64-127 -> Q2)
    const unsigned short* Qrow = Qb + (size_t)(b * T_ + q0 + mlane) * D2 + h * HDn;
    BFrag aQ[4];
#pragma unroll
    for (int kc = 0; kc < 4; ++kc) {
        aQ[kc].q[0] = *(const uint4*)&Qrow[kc * 32 + 16 * half];
        aQ[kc].q[1] = *(const uint4*)&Qrow[kc * 32 + 8 + 16 * half];
    }

    CFrag o1[8], o2[8];
    float mx1[8], mx2[8], l1[8], l2[8];
#pragma unroll
    for (int ct = 0; ct < 8; ++ct)
#pragma unroll
        for (int i = 0; i < 8; ++i) { o1[ct].f[i] = 0.0f; o2[ct].f[i] = 0.0f; }
#pragma unroll
    for (int r = 0; r < 8; ++r) { mx1[r] = -1e30f; mx2[r] = -1e30f; l1[r] = 0.0f; l2[r] = 0.0f; }

    unsigned short* P1 = &Plds[wid * 1024];
    unsigned short* P2 = P1 + 512;

    const int kend = q0 + 16;                   // causal: keys < kend
    for (int kb = 0; kb < kend; kb += 32) {
        // ---- scores: S = Q(16x64) @ K^T(64x32), per half (diff heads) ----
        CFrag s1[2], s2[2];
#pragma unroll
        for (int kt = 0; kt < 2; ++kt) {
            int key = kb + kt * 16 + mlane;
            const unsigned short* Krow = Kb + (size_t)(b * T_ + key) * D2 + h * HDn;
            v8f c1 = {0.f,0.f,0.f,0.f,0.f,0.f,0.f,0.f};
            v8f c2 = {0.f,0.f,0.f,0.f,0.f,0.f,0.f,0.f};
            BFrag bk;
#pragma unroll
            for (int kc = 0; kc < 2; ++kc) {                // channels 0..63
                const unsigned short* kp = &Krow[kc * 32 + 16 * half];
                bk.q[0] = *(const uint4*)kp;
                bk.q[1] = *(const uint4*)(kp + 8);
                c1 = __builtin_amdgcn_wmma_f32_16x16x32_bf16(
                        false, aQ[kc].v, false, bk.v, (short)0, c1, false, false);
            }
#pragma unroll
            for (int kc = 2; kc < 4; ++kc) {                // channels 64..127
                const unsigned short* kp = &Krow[kc * 32 + 16 * half];
                bk.q[0] = *(const uint4*)kp;
                bk.q[1] = *(const uint4*)(kp + 8);
                c2 = __builtin_amdgcn_wmma_f32_16x16x32_bf16(
                        false, aQ[kc].v, false, bk.v, (short)0, c2, false, false);
            }
            s1[kt].v = c1; s2[kt].v = c2;
        }
        // ---- scale + causal mask (data-independent) ----
#pragma unroll
        for (int kt = 0; kt < 2; ++kt) {
            int key = kb + kt * 16 + mlane;
#pragma unroll
            for (int r = 0; r < 8; ++r) {
                int row = q0 + r + 8 * half;
                bool ok = (key <= row);
                s1[kt].f[r] = ok ? s1[kt].f[r] * SCALE_QK : -1e30f;
                s2[kt].f[r] = ok ? s2[kt].f[r] * SCALE_QK : -1e30f;
            }
        }
        // ---- online softmax update (rows live in 16-lane halves) ----
#pragma unroll
        for (int r = 0; r < 8; ++r) {
            float rm1 = fmaxf(s1[0].f[r], s1[1].f[r]);
            float rm2 = fmaxf(s2[0].f[r], s2[1].f[r]);
#pragma unroll
            for (int msk = 1; msk < 16; msk <<= 1) {
                rm1 = fmaxf(rm1, __shfl_xor(rm1, msk, 32));
                rm2 = fmaxf(rm2, __shfl_xor(rm2, msk, 32));
            }
            float mn1 = fmaxf(mx1[r], rm1), mn2 = fmaxf(mx2[r], rm2);
            float a1 = __expf(mx1[r] - mn1), a2 = __expf(mx2[r] - mn2);
            float p10 = __expf(s1[0].f[r] - mn1), p11 = __expf(s1[1].f[r] - mn1);
            float p20 = __expf(s2[0].f[r] - mn2), p21 = __expf(s2[1].f[r] - mn2);
            float rs1 = p10 + p11, rs2 = p20 + p21;
#pragma unroll
            for (int msk = 1; msk < 16; msk <<= 1) {
                rs1 += __shfl_xor(rs1, msk, 32);
                rs2 += __shfl_xor(rs2, msk, 32);
            }
            l1[r] = l1[r] * a1 + rs1;  mx1[r] = mn1;
            l2[r] = l2[r] * a2 + rs2;  mx2[r] = mn2;
#pragma unroll
            for (int ct = 0; ct < 8; ++ct) { o1[ct].f[r] *= a1; o2[ct].f[r] *= a2; }
            // P -> LDS (row-major 16x32 bf16) for C-layout -> A-layout shuffle
            int prow = r + 8 * half;
            P1[prow * 32 +      mlane] = f2bf(p10);
            P1[prow * 32 + 16 + mlane] = f2bf(p11);
            P2[prow * 32 +      mlane] = f2bf(p20);
            P2[prow * 32 + 16 + mlane] = f2bf(p21);
        }
        asm volatile("s_wait_dscnt 0x0" ::: "memory");   // per-wave LDS RAW fence
        // ---- P as A operand (2x b128 per matrix) ----
        BFrag aP1, aP2;
        {
            const unsigned short* pb1 = &P1[mlane * 32 + 16 * half];
            const unsigned short* pb2 = &P2[mlane * 32 + 16 * half];
            aP1.q[0] = *(const uint4*)pb1;  aP1.q[1] = *(const uint4*)(pb1 + 8);
            aP2.q[0] = *(const uint4*)pb2;  aP2.q[1] = *(const uint4*)(pb2 + 8);
        }
        // ---- O += P @ V over 8 channel tiles (V^T gives contiguous key runs)
#pragma unroll
        for (int ct = 0; ct < 8; ++ct) {
            const unsigned short* vb =
                Vt + (size_t)(h * HDn + ct * 16 + mlane) * BT + (size_t)b * T_
                   + kb + 16 * half;
            BFrag bv;
            bv.q[0] = *(const uint4*)vb;
            bv.q[1] = *(const uint4*)(vb + 8);
            o1[ct].v = __builtin_amdgcn_wmma_f32_16x16x32_bf16(
                false, aP1.v, false, bv.v, (short)0, o1[ct].v, false, false);
            o2[ct].v = __builtin_amdgcn_wmma_f32_16x16x32_bf16(
                false, aP2.v, false, bv.v, (short)0, o2[ct].v, false, false);
        }
    }

    // ---- epilogue: O = O1/l1 - sigmoid(lambda)*O2/l2 ----
    const float lam = 1.0f / (1.0f + __expf(-lambda_init[h]));
    unsigned short* Orow =
        Ob + (size_t)(b * T_ + q0 + 8 * half) * D2 + h * HDn + mlane;
#pragma unroll
    for (int r = 0; r < 8; ++r) {
        float inv1 = 1.0f / l1[r];
        float inv2g = lam / l2[r];
#pragma unroll
        for (int ct = 0; ct < 8; ++ct) {
            float vv = o1[ct].f[r] * inv1 - o2[ct].f[r] * inv2g;
            Orow[(size_t)r * D2 + ct * 16] = f2bf(vv);
        }
    }
}

// ---------------------------------------------------------------------------
// Host launcher
// ---------------------------------------------------------------------------
extern "C" void kernel_launch(void* const* d_in, const int* in_sizes, int n_in,
                              void* d_out, int out_size, void* d_ws, size_t ws_size,
                              hipStream_t stream) {
    constexpr int Bc = 2, T = 2048, D = 2048, Hn = 16, HDn = 128;
    constexpr int M = Bc * T;                   // 4096

    const float* x   = (const float*)d_in[0];
    const float* Wq  = (const float*)d_in[1];
    const float* Wk  = (const float*)d_in[2];
    const float* Wv  = (const float*)d_in[3];
    const float* Wo  = (const float*)d_in[4];
    const float* lam = (const float*)d_in[5];

    char* ws = (char*)d_ws;
    const size_t MB = 1024 * 1024;
    unsigned short* xb   = (unsigned short*)(ws + 0);
    unsigned short* Wqb  = (unsigned short*)(ws + 16 * MB);
    unsigned short* Wkb  = (unsigned short*)(ws + 24 * MB);
    unsigned short* Wvb  = (unsigned short*)(ws + 32 * MB);
    unsigned short* Wob  = (unsigned short*)(ws + 40 * MB);
    unsigned short* Qb   = (unsigned short*)(ws + 48 * MB);
    unsigned short* Kbuf = (unsigned short*)(ws + 64 * MB);
    unsigned short* Vt   = (unsigned short*)(ws + 80 * MB);
    unsigned short* Ab   = (unsigned short*)(ws + 96 * MB);

    // 1) fp32 -> bf16 (vectorized x4)
    int nx4 = (M * D) / 4, nw4 = (D * D) / 4;
    cvt_f32_bf16<<<(nx4 + 255) / 256, 256, 0, stream>>>((const float4*)x,  (uint2*)xb,  nx4);
    cvt_f32_bf16<<<(nw4 + 255) / 256, 256, 0, stream>>>((const float4*)Wq, (uint2*)Wqb, nw4);
    cvt_f32_bf16<<<(nw4 + 255) / 256, 256, 0, stream>>>((const float4*)Wk, (uint2*)Wkb, nw4);
    cvt_f32_bf16<<<(nw4 + 255) / 256, 256, 0, stream>>>((const float4*)Wv, (uint2*)Wvb, nw4);
    cvt_f32_bf16<<<(nw4 + 255) / 256, 256, 0, stream>>>((const float4*)Wo, (uint2*)Wob, nw4);

    // 2) projections (V stored transposed for attention PV operand)
    dim3 ggrid(D / BN, M / BM);                 // (16, 32)
    gemm_bf16<M, D, D, false, false><<<ggrid, 256, 0, stream>>>(xb, Wqb, Qb);
    gemm_bf16<M, D, D, false, false><<<ggrid, 256, 0, stream>>>(xb, Wkb, Kbuf);
    gemm_bf16<M, D, D, false, true ><<<ggrid, 256, 0, stream>>>(xb, Wvb, Vt);

    // 3) differential attention
    dim3 agrid(Bc * Hn, T / 128);               // (32, 16)
    diff_attn<T, Hn, HDn, Bc><<<agrid, 256, 0, stream>>>(Qb, Kbuf, Vt, lam, Ab);

    // 4) output projection -> fp32 d_out
    gemm_bf16<M, D, D, true, false><<<ggrid, 256, 0, stream>>>(Ab, Wob, d_out);
}
